// MultiHeadAttention_13993003450746
// MI455X (gfx1250) — matrix operand order repaired and checked by reading, not compiled, not run on hardware
//
#include <hip/hip_runtime.h>
#include <hip/hip_bf16.h>

// MI455X (gfx1250) multi-head attention:
//   3x projection GEMM (f32->f16, WMMA f32_16x16x32_f16)  -> qh/kh row-major, vt transposed
//   1x flash-attention: TDM double-buffered K/V tiles in LDS, WMMA QK^T +
//       online softmax (permlane16 butterflies) + WMMA P*V
//   1x output GEMM (f16 in via async global->LDS copy, f32 out + bias)
// Workspace: 4 x 16 MiB f16 buffers = 64 MiB of d_ws.

typedef _Float16 f16;
typedef __attribute__((ext_vector_type(16))) _Float16 v16h;
typedef __attribute__((ext_vector_type(8)))  _Float16 h8;
typedef __attribute__((ext_vector_type(4)))  _Float16 h4;
typedef __attribute__((ext_vector_type(8)))  float    v8f;
typedef __attribute__((ext_vector_type(4)))  float    f4;
typedef __attribute__((ext_vector_type(4)))  int      i4;
typedef __attribute__((ext_vector_type(4)))  unsigned u4;
typedef __attribute__((ext_vector_type(8)))  int      i8v;
typedef __attribute__((address_space(1)))    i4       gi4;   // global v4i
typedef __attribute__((address_space(3)))    i4       li4;   // LDS v4i

#define D_MODEL 1024
#define SEQ     2048
#define BATCH   4
#define HEADS   16
#define DK      64

#if defined(__has_builtin)
#  if __has_builtin(__builtin_amdgcn_permlane16)
#    define HAVE_PERMLANE16 1
#  endif
#  if __has_builtin(__builtin_amdgcn_global_load_async_to_lds_b128)
#    define HAVE_ASYNC_LDS 1
#  endif
#  if __has_builtin(__builtin_amdgcn_tensor_load_to_lds)
#    define HAVE_TDM 1
#    if __clang_major__ >= 23
#      define TDM_6ARG 1   // amdgpu-toolchain: (g0, g1, g2, g3, g4, cpol)
#    else
#      define TDM_6ARG 0   // ROCm 7.2: (g0, g1, g2, g3, cpol)
#    endif
#  endif
#endif
#ifndef HAVE_PERMLANE16
#  define HAVE_PERMLANE16 0
#endif
#ifndef HAVE_ASYNC_LDS
#  define HAVE_ASYNC_LDS 0
#endif
#ifndef HAVE_TDM
#  define HAVE_TDM 0
#endif

static __device__ __forceinline__ v16h cat16(h8 lo, h8 hi) {
  return __builtin_shufflevector(lo, hi, 0,1,2,3,4,5,6,7,8,9,10,11,12,13,14,15);
}
static __device__ __forceinline__ v8f zero8() {
  v8f z;
#pragma unroll
  for (int i = 0; i < 8; ++i) z[i] = 0.0f;
  return z;
}
static __device__ __forceinline__ v8f wmma16(v16h a, v16h b, v8f c) {
  // D = A(16x32 f16) * B(32x16 f16) + C(16x16 f32)
  return __builtin_amdgcn_wmma_f32_16x16x32_f16(false, a, false, b, (short)0, c,
                                                false, false);
}

#if HAVE_PERMLANE16
static __device__ __forceinline__ float pl16(float v, unsigned lo, unsigned hi) {
  unsigned u = __builtin_bit_cast(unsigned, v);
  u = __builtin_amdgcn_permlane16(u, u, lo, hi, false, false);
  return __builtin_bit_cast(float, u);
}
#endif
// butterfly reduction over the 16-lane N-group (lanes i <-> i^m, m=1,2,4,8)
static __device__ __forceinline__ float red_max16(float v) {
#if HAVE_PERMLANE16
  v = fmaxf(v, pl16(v, 0x67452301u, 0xEFCDAB89u));  // xor 1
  v = fmaxf(v, pl16(v, 0x54761032u, 0xDCFE98BAu));  // xor 2
  v = fmaxf(v, pl16(v, 0x32107654u, 0xBA98FEDCu));  // xor 4
  v = fmaxf(v, pl16(v, 0xFEDCBA98u, 0x76543210u));  // xor 8
#else
#pragma unroll
  for (int off = 1; off < 16; off <<= 1) v = fmaxf(v, __shfl_xor(v, off, 32));
#endif
  return v;
}
static __device__ __forceinline__ float red_sum16(float v) {
#if HAVE_PERMLANE16
  v += pl16(v, 0x67452301u, 0xEFCDAB89u);
  v += pl16(v, 0x54761032u, 0xDCFE98BAu);
  v += pl16(v, 0x32107654u, 0xBA98FEDCu);
  v += pl16(v, 0xFEDCBA98u, 0x76543210u);
#else
#pragma unroll
  for (int off = 1; off < 16; off <<= 1) v += __shfl_xor(v, off, 32);
#endif
  return v;
}

#if HAVE_TDM
static __device__ __forceinline__ unsigned lds_off(const void* p) {
  return (unsigned)(unsigned long long)(__attribute__((address_space(3))) const char*)p;
}
// Build a 2D tensor-DMA descriptor (ISA section 8.3/8.4) and issue the load.
// data_size=2B; LDS padding: pad_amount DWORD-code after each pad_interval code.
static __device__ __forceinline__ void tdm_load_tile(
    unsigned lds_byte_off, unsigned long long gaddr,
    unsigned tile_w, unsigned tile_h,
    unsigned tensor_w, unsigned tensor_h, unsigned row_stride,
    unsigned pad_interval_code, unsigned pad_amount_code) {
  u4 g0;
  g0[0] = 1u;                                       // count=1, user mode
  g0[1] = lds_byte_off;                             // lds_addr
  g0[2] = (unsigned)(gaddr & 0xFFFFFFFFull);        // global_addr[31:0]
  g0[3] = ((unsigned)(gaddr >> 32) & 0x1FFFFFFu) | (2u << 30);  // addr[56:32]|type=2
  i8v g1;
  g1[0] = (int)((1u << 16) | (1u << 20) |
                (pad_interval_code << 22) | (pad_amount_code << 25));
  g1[1] = (int)((tensor_w & 0xFFFFu) << 16);                          // dim0[15:0]
  g1[2] = (int)(((tensor_w >> 16) & 0xFFFFu) | ((tensor_h & 0xFFFFu) << 16));
  g1[3] = (int)(((tensor_h >> 16) & 0xFFFFu) | ((tile_w & 0xFFFFu) << 16));
  g1[4] = (int)(tile_h & 0xFFFFu);                                    // tile_dim1
  g1[5] = (int)row_stride;                                            // dim0_stride
  g1[6] = 0;
  g1[7] = 0;
  i4 gz; gz[0] = 0; gz[1] = 0; gz[2] = 0; gz[3] = 0;
#if TDM_6ARG
  i8v g4z;
#pragma unroll
  for (int i = 0; i < 8; ++i) g4z[i] = 0;
  __builtin_amdgcn_tensor_load_to_lds(g0, g1, gz, gz, g4z, 0);
#else
  __builtin_amdgcn_tensor_load_to_lds(g0, g1, gz, gz, 0);
#endif
}
#endif

// ---------------------------------------------------------------------------
// GEMM: Out(M=8192, N=1024) = X(M,1024) @ W(1024,1024)^T + bias
// MODE 0: store f16 per-head row-major  out[b][h][t][d]
// MODE 1: store f16 per-head transposed out[b][h][d][t]   (packed 16B stores)
// MODE 2: store f32 plain row-major     out[m][n]
// WG tile 128x128, 8 waves (2x4), wave tile 64x32 (4x2 fragments), K step 32.
// ---------------------------------------------------------------------------
template <typename TIN, int MODE>
__global__ __launch_bounds__(256) void gemm128(const TIN*  __restrict__ X,
                                               const float* __restrict__ W,
                                               const float* __restrict__ bias,
                                               void* __restrict__ OutP) {
  constexpr int KD = D_MODEL;
  __shared__ f16 Asm[128][40];   // 80B row stride: 16B-aligned sub-chunks
  __shared__ f16 Bsm[128][40];   // Bsm[n][k] = W[n0+n][kk+k]

  const int tid  = threadIdx.x;
  const int lane = tid & 31;
  const int wv   = tid >> 5;
  const int wm   = wv >> 2;            // 0..1
  const int wn   = wv & 3;             // 0..3
  const int half = lane >> 4;
  const int l16  = lane & 15;
  const int m0 = blockIdx.x * 128;
  const int n0 = blockIdx.y * 128;

  v8f acc[4][2];
#pragma unroll
  for (int i = 0; i < 4; ++i)
#pragma unroll
    for (int j = 0; j < 2; ++j) acc[i][j] = zero8();

  const int c4 = tid & 7;              // chunk of 4 floats within k-step
  const int r4 = tid >> 3;             // 0..31

  for (int kk = 0; kk < KD; kk += 32) {
    // ---- stage B tile (weights always f32, convert to f16) ----
#pragma unroll
    for (int rr = r4; rr < 128; rr += 32) {
      f4 wv4 = *reinterpret_cast<const f4*>(W + (size_t)(n0 + rr) * KD + kk + 4 * c4);
      h4 wh;
#pragma unroll
      for (int i = 0; i < 4; ++i) wh[i] = (f16)wv4[i];
      *reinterpret_cast<h4*>(&Bsm[rr][4 * c4]) = wh;
    }
    // ---- stage A tile ----
    if constexpr (sizeof(TIN) == 4) {
#pragma unroll
      for (int rr = r4; rr < 128; rr += 32) {
        f4 xv = *reinterpret_cast<const f4*>((const float*)X + (size_t)(m0 + rr) * KD + kk + 4 * c4);
        h4 xh;
#pragma unroll
        for (int i = 0; i < 4; ++i) xh[i] = (f16)xv[i];
        *reinterpret_cast<h4*>(&Asm[rr][4 * c4]) = xh;
      }
    } else {
      const int c8 = tid & 3, r8 = tid >> 2;  // 0..63
#pragma unroll
      for (int rr = r8; rr < 128; rr += 64) {
#if HAVE_ASYNC_LDS
        // gfx1250 async global->LDS copy, tracked by ASYNCcnt.
        __builtin_amdgcn_global_load_async_to_lds_b128(
            (gi4*)(uintptr_t)((const f16*)X + (size_t)(m0 + rr) * KD + kk + 8 * c8),
            (li4*)&Asm[rr][8 * c8],
            0, 0);
#else
        h8 xv = *reinterpret_cast<const h8*>((const f16*)X + (size_t)(m0 + rr) * KD + kk + 8 * c8);
        *reinterpret_cast<h8*>(&Asm[rr][8 * c8]) = xv;
#endif
      }
#if HAVE_ASYNC_LDS
      asm volatile("s_wait_asynccnt 0x0" ::: "memory");
#endif
    }
    if (kk + 32 < KD) {                // gfx1250 global_prefetch_b8
      __builtin_prefetch((const char*)X + ((size_t)(m0 + r4) * KD + kk + 32) * sizeof(TIN), 0, 0);
      __builtin_prefetch(W + (size_t)(n0 + r4) * KD + kk + 32, 0, 0);
    }
    __syncthreads();

    // ---- fragments + WMMA ----
    v16h bfr[2];
#pragma unroll
    for (int nt = 0; nt < 2; ++nt) {
      const f16* bp = &Bsm[wn * 32 + nt * 16 + l16][half * 8];
      bfr[nt] = cat16(*reinterpret_cast<const h8*>(bp),
                      *reinterpret_cast<const h8*>(bp + 16));
    }
#pragma unroll
    for (int mt = 0; mt < 4; ++mt) {
      const f16* ap = &Asm[wm * 64 + mt * 16 + l16][0];
      v16h afr = cat16(*reinterpret_cast<const h8*>(ap + half * 8),
                       *reinterpret_cast<const h8*>(ap + 16 + half * 8));
#pragma unroll
      for (int nt = 0; nt < 2; ++nt)
        acc[mt][nt] = wmma16(afr, bfr[nt], acc[mt][nt]);
    }
    __syncthreads();
  }

  // ---- epilogue: bias + store ----
#pragma unroll
  for (int mt = 0; mt < 4; ++mt)
#pragma unroll
    for (int nt = 0; nt < 2; ++nt) {
      const int n   = n0 + wn * 32 + nt * 16 + l16;
      const float bb = bias[n];
      const int mr0 = m0 + wm * 64 + mt * 16 + half * 8;
      if constexpr (MODE == 2) {
        float* Out = reinterpret_cast<float*>(OutP);
#pragma unroll
        for (int r = 0; r < 8; ++r)
          Out[(size_t)(mr0 + r) * D_MODEL + n] = acc[mt][nt][r] + bb;
      } else if constexpr (MODE == 0) {
        f16* Out = reinterpret_cast<f16*>(OutP);
        const int hh = n >> 6, d = n & (DK - 1);
#pragma unroll
        for (int r = 0; r < 8; ++r) {
          const int m = mr0 + r;
          const int bi = m >> 11, t = m & (SEQ - 1);
          Out[(((size_t)bi * HEADS + hh) * SEQ + t) * DK + d] = (f16)(acc[mt][nt][r] + bb);
        }
      } else {  // MODE 1: transposed per-head, lane's 8 rows are contiguous t
        f16* Out = reinterpret_cast<f16*>(OutP);
        const int hh = n >> 6, d = n & (DK - 1);
        const int bi = mr0 >> 11, t = mr0 & (SEQ - 1);
        h8 pk;
#pragma unroll
        for (int r = 0; r < 8; ++r) pk[r] = (f16)(acc[mt][nt][r] + bb);
        *reinterpret_cast<h8*>(&Out[(((size_t)bi * HEADS + hh) * DK + d) * SEQ + t]) = pk;
      }
    }
}

// ---------------------------------------------------------------------------
// Flash attention: one wave per 16 q rows, 64-wide key tiles.
// Q: (b,h,t,64) f16; K: (b,h,t,64) f16; Vt: (b,h,64,t) f16.
// With TDM: wave 0 DMA-stages K/V tiles (64x64 f16, 144B padded rows) into
// double-buffered LDS; DMA of tile i+1 overlaps compute of tile i.
// ---------------------------------------------------------------------------
__global__ __launch_bounds__(256) void attn_fa(const f16* __restrict__ Q,
                                               const f16* __restrict__ K,
                                               const f16* __restrict__ Vt,
                                               const unsigned char* __restrict__ mask,
                                               f16* __restrict__ O) {
#if HAVE_TDM
  __shared__ f16 Ksm[2][64][72];  // 144B rows: TDM pad 16B after each 128B row
  __shared__ f16 Vsm[2][64][72];
#endif
  __shared__ f16 Pl[8][16][72];   // per-wave P (16x64) transpose scratch
  const int tid  = threadIdx.x;
  const int lane = tid & 31, wv = tid >> 5;
  const int half = lane >> 4, l16 = lane & 15;
  const int bh = blockIdx.y;
  const int bi = bh >> 4;             // batch
  const int hh = bh & (HEADS - 1);    // head
  const int q0 = blockIdx.x * 128 + wv * 16;

  const f16* Kbase = K  + (size_t)bh * SEQ * DK;
  const f16* Vbase = Vt + (size_t)bh * DK * SEQ;

  // Q A-fragments for dk 0..31 / 32..63; pre-scale by 1/sqrt(64)=0.125 (exact)
  const size_t qb = ((size_t)bh * SEQ + q0 + l16) * DK;
  v16h a0 = cat16(*reinterpret_cast<const h8*>(Q + qb + half * 8),
                  *reinterpret_cast<const h8*>(Q + qb + 16 + half * 8));
  v16h a1 = cat16(*reinterpret_cast<const h8*>(Q + qb + 32 + half * 8),
                  *reinterpret_cast<const h8*>(Q + qb + 48 + half * 8));
#pragma unroll
  for (int i = 0; i < 16; ++i) {
    a0[i] = a0[i] * (f16)0.125f;
    a1[i] = a1[i] * (f16)0.125f;
  }

  v8f o[4];
#pragma unroll
  for (int nb = 0; nb < 4; ++nb) o[nb] = zero8();
  float mrow[8], lrow[8];
#pragma unroll
  for (int r = 0; r < 8; ++r) { mrow[r] = -__builtin_inff(); lrow[r] = 0.0f; }

#if HAVE_TDM
  if (wv == 0) {  // prologue: DMA tile 0 into buffer 0 (TENSORcnt += 2)
    tdm_load_tile(lds_off(&Ksm[0][0][0]), (unsigned long long)(uintptr_t)Kbase,
                  64, 64, DK, (unsigned)(BATCH * HEADS * SEQ), DK, 4, 3);
    tdm_load_tile(lds_off(&Vsm[0][0][0]), (unsigned long long)(uintptr_t)Vbase,
                  64, 64, SEQ, (unsigned)(BATCH * HEADS * DK), SEQ, 4, 3);
  }
#endif

  for (int it = 0; it < SEQ / 64; ++it) {
    const int kt = it * 64;
#if HAVE_TDM
    const int cur = it & 1;
    if (wv == 0) __builtin_amdgcn_s_wait_tensorcnt(0);  // tile it arrived
    __syncthreads();  // publish buf[cur]; all reads of buf[cur^1] retired
    if (wv == 0 && it + 1 < SEQ / 64) {
      const int kn = kt + 64;
      tdm_load_tile(lds_off(&Ksm[cur ^ 1][0][0]),
                    (unsigned long long)(uintptr_t)(Kbase + (size_t)kn * DK),
                    64, 64, DK, (unsigned)(BATCH * HEADS * SEQ), DK, 4, 3);
      tdm_load_tile(lds_off(&Vsm[cur ^ 1][0][0]),
                    (unsigned long long)(uintptr_t)(Vbase + kn),
                    64, 64, SEQ, (unsigned)(BATCH * HEADS * DK), SEQ, 4, 3);
    }
#endif

    // ---- S fragments: 4 column groups of 16 keys ----
    v8f s[4];
#pragma unroll
    for (int c = 0; c < 4; ++c) {
#if HAVE_TDM
      const f16* kp = &Ksm[cur][c * 16 + l16][half * 8];
#else
      const f16* kp = Kbase + (size_t)(kt + c * 16 + l16) * DK + half * 8;
#endif
      v16h blo = cat16(*(const h8*)(kp),      *(const h8*)(kp + 16));
      v16h bhi = cat16(*(const h8*)(kp + 32), *(const h8*)(kp + 48));
      s[c] = wmma16(a0, blo, zero8());
      s[c] = wmma16(a1, bhi, s[c]);
      // branchless mask (all EXEC lanes stay active for WMMA)
      const float mb = mask[(size_t)bi * SEQ + kt + c * 16 + l16] ? 0.0f : -1e30f;
#pragma unroll
      for (int r = 0; r < 8; ++r) s[c][r] += mb;
    }

    // ---- online softmax ----
    float nm[8], fac[8];
#pragma unroll
    for (int r = 0; r < 8; ++r) {
      float v = fmaxf(fmaxf(s[0][r], s[1][r]), fmaxf(s[2][r], s[3][r]));
      v = red_max16(v);
      nm[r]  = fmaxf(mrow[r], v);
      fac[r] = __expf(mrow[r] - nm[r]);   // first tile: exp(-inf)=0
      mrow[r] = nm[r];
    }
#pragma unroll
    for (int r = 0; r < 8; ++r) {
#pragma unroll
      for (int c = 0; c < 4; ++c) s[c][r] = __expf(s[c][r] - nm[r]);
      float rs = (s[0][r] + s[1][r]) + (s[2][r] + s[3][r]);
      rs = red_sum16(rs);
      lrow[r] = lrow[r] * fac[r] + rs;
    }
#pragma unroll
    for (int nb = 0; nb < 4; ++nb)
#pragma unroll
      for (int r = 0; r < 8; ++r) o[nb][r] *= fac[r];

    // ---- P: C-layout f32 -> A-layout f16 via per-wave LDS scratch ----
#pragma unroll
    for (int c = 0; c < 4; ++c)
#pragma unroll
      for (int r = 0; r < 8; ++r)
        Pl[wv][r + half * 8][c * 16 + l16] = (f16)s[c][r];
    asm volatile("s_wait_dscnt 0x0" ::: "memory");
    const f16* pr = &Pl[wv][l16][0];
    v16h ap0 = cat16(*reinterpret_cast<const h8*>(pr + half * 8),
                     *reinterpret_cast<const h8*>(pr + 16 + half * 8));
    v16h ap1 = cat16(*reinterpret_cast<const h8*>(pr + 32 + half * 8),
                     *reinterpret_cast<const h8*>(pr + 48 + half * 8));
    asm volatile("" ::: "memory");  // keep next-iter DS stores after these loads

    // ---- O += P * V ----
#pragma unroll
    for (int nb = 0; nb < 4; ++nb) {
#if HAVE_TDM
      const f16* vp = &Vsm[cur][nb * 16 + l16][half * 8];
#else
      const f16* vp = Vbase + (size_t)(nb * 16 + l16) * SEQ + kt + half * 8;
#endif
      v16h bv0 = cat16(*(const h8*)(vp),      *(const h8*)(vp + 16));
      v16h bv1 = cat16(*(const h8*)(vp + 32), *(const h8*)(vp + 48));
      o[nb] = wmma16(ap0, bv0, o[nb]);
      o[nb] = wmma16(ap1, bv1, o[nb]);
    }
#if !HAVE_TDM
    if (kt + 64 < SEQ)
      __builtin_prefetch(Kbase + (size_t)(kt + 64 + l16) * DK, 0, 0);
#endif
  }

  // ---- normalize and store attn output row-major (b, t, h*64+d) f16 ----
#pragma unroll
  for (int r = 0; r < 8; ++r) {
    const float inv = 1.0f / lrow[r];
    const int t = q0 + half * 8 + r;
#pragma unroll
    for (int nb = 0; nb < 4; ++nb) {
      const int col = hh * DK + nb * 16 + l16;
      O[((size_t)bi * SEQ + t) * D_MODEL + col] = (f16)(o[nb][r] * inv);
    }
  }
}

// ---------------------------------------------------------------------------
extern "C" void kernel_launch(void* const* d_in, const int* in_sizes, int n_in,
                              void* d_out, int out_size, void* d_ws, size_t ws_size,
                              hipStream_t stream) {
  const float* q   = (const float*)d_in[0];
  const float* k   = (const float*)d_in[1];
  const float* v   = (const float*)d_in[2];
  const unsigned char* mask = (const unsigned char*)d_in[3];
  const float* w_q = (const float*)d_in[4];
  const float* b_q = (const float*)d_in[5];
  const float* w_k = (const float*)d_in[6];
  const float* b_k = (const float*)d_in[7];
  const float* w_v = (const float*)d_in[8];
  const float* b_v = (const float*)d_in[9];
  const float* w_o = (const float*)d_in[10];
  const float* b_o = (const float*)d_in[11];

  const size_t elems = (size_t)BATCH * SEQ * D_MODEL;  // 8 Mi f16 per buffer
  f16* qh = (f16*)d_ws;          // (b,h,t,64)
  f16* kh = qh + elems;          // (b,h,t,64)
  f16* vt = kh + elems;          // (b,h,64,t)
  f16* at = vt + elems;          // (b,t,1024)

  dim3 blk(256);
  dim3 gg(BATCH * SEQ / 128, D_MODEL / 128);   // 64 x 8
  gemm128<float, 0><<<gg, blk, 0, stream>>>(q, w_q, b_q, qh);
  gemm128<float, 0><<<gg, blk, 0, stream>>>(k, w_k, b_k, kh);
  gemm128<float, 1><<<gg, blk, 0, stream>>>(v, w_v, b_v, vt);

  dim3 ga(SEQ / 128, BATCH * HEADS);           // 16 x 64
  attn_fa<<<ga, blk, 0, stream>>>(qh, kh, vt, mask, at);

  gemm128<f16, 2><<<gg, blk, 0, stream>>>(at, w_o, b_o, d_out);

  (void)in_sizes; (void)n_in; (void)out_size; (void)ws_size;
}
